// MultiHeadAttention_88699664597743
// MI455X (gfx1250) — matrix operand order repaired and checked by reading, not compiled, not run on hardware
//
#include <hip/hip_runtime.h>
#include <hip/hip_bf16.h>

// MHA: B=8, L=1024, E=1024, H=16, D=64
#define BB 8
#define LL 1024
#define EE 1024
#define HH 16
#define DD 64

typedef __attribute__((ext_vector_type(16))) _Float16 v16h;
typedef __attribute__((ext_vector_type(8)))  _Float16 v8h;
typedef __attribute__((ext_vector_type(8)))  float    v8f;

__device__ __forceinline__ v8f wmma_f16(v16h a, v16h b, v8f c) {
    return __builtin_amdgcn_wmma_f32_16x16x32_f16(false, a, false, b, (short)0, c, false, false);
}

// ---- fragment loaders per CDNA5 ISA 7.12.2 layouts (wave32), vectorized (2 x b128 per lane) ----
// A (16x32 f16) from row-major [M x K]: a[0..7] = K 8*hl+0..7 ; a[8..15] = K 16+8*hl+0..7
__device__ __forceinline__ v16h load_a_f16(const _Float16* base, int ld, int lane) {
    int m = lane & 15, hl = lane >> 4;
    const _Float16* p = base + (size_t)m * ld + 8 * hl;
    v8h c0 = *(const v8h*)(p);
    v8h c1 = *(const v8h*)(p + 16);
    return __builtin_shufflevector(c0, c1, 0,1,2,3,4,5,6,7,8,9,10,11,12,13,14,15);
}
__device__ __forceinline__ v16h load_a_f32(const float* base, int ld, int lane) {
    int m = lane & 15, hl = lane >> 4;
    const float* p = base + (size_t)m * ld + 8 * hl;
    v8f c0 = *(const v8f*)(p);
    v8f c1 = *(const v8f*)(p + 16);
    v16h a;
#pragma unroll
    for (int i = 0; i < 8; ++i) { a[i] = (_Float16)c0[i]; a[8 + i] = (_Float16)c1[i]; }
    return a;
}
// B (32x16 f16) where the K-dim is contiguous in memory: B[kk][n] = base[n*ld + kk]
__device__ __forceinline__ v16h load_b_colT(const _Float16* base, int ld, int lane) {
    int n = lane & 15, hl = lane >> 4;
    const _Float16* p = base + (size_t)n * ld + 8 * hl;
    v8h c0 = *(const v8h*)(p);
    v8h c1 = *(const v8h*)(p + 16);
    return __builtin_shufflevector(c0, c1, 0,1,2,3,4,5,6,7,8,9,10,11,12,13,14,15);
}

// ---- kernel 0: weight conversion (transposed for contiguous B-fragments) + key bias ----
__global__ __launch_bounds__(256) void mha_prep_kernel(
    const float* __restrict__ HL, const float* __restrict__ WO, const int* __restrict__ KM,
    _Float16* __restrict__ whT, _Float16* __restrict__ woT, float* __restrict__ kbias) {
    int i = blockIdx.x * 256 + threadIdx.x;
    if (i < HH * DD * DD) {          // whT[h][e_out][d_in] = HL[h][d_in][e_out]
        int h = i >> 12, rem = i & 4095, e = rem >> 6, d = rem & 63;
        whT[i] = (_Float16)HL[h * 4096 + d * 64 + e];
    }
    if (i < EE * EE) {               // woT[e_out][e_in] = WO[e_in][e_out]
        int nn = i >> 10, k = i & 1023;
        woT[i] = (_Float16)WO[k * EE + nn];
    }
    if (i < BB * LL) kbias[i] = (KM[i] != 0) ? 0.0f : -1e10f;
}

// ---- kernel 1: per-head projection. qh/kh row-major [B,H,L,D]; V stored transposed [B,H,D,L].
__global__ __launch_bounds__(256) void mha_proj_kernel(
    const float* __restrict__ Q, const float* __restrict__ K, const float* __restrict__ V,
    const _Float16* __restrict__ whT,
    _Float16* __restrict__ qh, _Float16* __restrict__ kh, _Float16* __restrict__ vhT) {
    int lane = threadIdx.x & 31;
    int wid  = blockIdx.x * 8 + (threadIdx.x >> 5);
    int lt = wid & 63, h = (wid >> 6) & 15, b = (wid >> 10) & 7, t = wid >> 13;
    const float* X = (t == 0) ? Q : (t == 1) ? K : V;
    int l0 = lt * 16;
    v8f c[4] = {};
#pragma unroll
    for (int ks = 0; ks < 2; ++ks) {
        v16h a = load_a_f32(X + (size_t)(b * LL + l0) * EE + h * DD + ks * 32, EE, lane);
#pragma unroll
        for (int nt = 0; nt < 4; ++nt) {
            v16h bb = load_b_colT(whT + h * DD * DD + (nt * 16) * DD + ks * 32, DD, lane);
            c[nt] = wmma_f16(a, bb, c[nt]);
        }
    }
    int n = lane & 15, hl = lane >> 4;
    if (t < 2) {   // row-major store [B,H,L,D]
        _Float16* dst = (t == 0) ? qh : kh;
#pragma unroll
        for (int nt = 0; nt < 4; ++nt)
#pragma unroll
            for (int r = 0; r < 8; ++r) {
                int m = r + 8 * hl;
                dst[((size_t)((b * HH + h) * LL) + l0 + m) * DD + nt * 16 + n] = (_Float16)c[nt][r];
            }
    } else {       // transposed store [B,H,D,L]: one b128 store per (lane, nt)
#pragma unroll
        for (int nt = 0; nt < 4; ++nt) {
            v8h pk;
#pragma unroll
            for (int r = 0; r < 8; ++r) pk[r] = (_Float16)c[nt][r];
            *(v8h*)(vhT + ((size_t)((b * HH + h) * DD) + nt * 16 + n) * LL + l0 + 8 * hl) = pk;
        }
    }
}

// ---- kernel 2: flash attention, block-cooperative. One block = 8 waves sharing LDS-staged
// 64-key K/V chunks (double buffered); each wave owns 2 query tiles (32 queries).
// Grid: 4 blocks per (b,h) -> K/V read from L2 only 4x total instead of 64x.
__global__ __launch_bounds__(256) void mha_attn_kernel(
    const _Float16* __restrict__ qh, const _Float16* __restrict__ kh, const _Float16* __restrict__ vhT,
    const float* __restrict__ kbias, const int* __restrict__ QMask, _Float16* __restrict__ o16) {
    __shared__ _Float16 ksm[2][64 * DD];   // [keys 64][d 64], row-major, 8KB per buffer
    __shared__ _Float16 vsm[2][DD * 64];   // [d 64][keys 64] (V^T), 8KB per buffer

    int tid  = threadIdx.x;
    int lane = tid & 31;
    int w    = tid >> 5;                   // wave 0..7
    int bh   = blockIdx.x >> 2;            // (b*16+h)
    int qq   = blockIdx.x & 3;             // query quadrant (256 queries)
    int b    = bh >> 4, h = bh & 15;
    int n = lane & 15, hl = lane >> 4;
    const _Float16* qbh  = qh  + (size_t)bh * LL * DD;
    const _Float16* kbh  = kh  + (size_t)bh * LL * DD;
    const _Float16* vbhT = vhT + (size_t)bh * DD * LL;
    int q0[2] = { qq * 256 + w * 32, qq * 256 + w * 32 + 16 };

    // Q^T B-fragments (kk = d), resident for the whole kernel
    v16h bq[2][2];
#pragma unroll
    for (int qi = 0; qi < 2; ++qi)
#pragma unroll
        for (int dc = 0; dc < 2; ++dc)
            bq[qi][dc] = load_b_colT(qbh + (size_t)q0[qi] * DD + dc * 32, DD, lane);

    v8f o[2][4] = {};
    float m_run[2] = { -3.0e38f, -3.0e38f };
    float l_run[2] = { 0.0f, 0.0f };
    const float scale = 0.125f; // 1/sqrt(64)

    int s1i = tid + 256;
    // stage chunk 0 directly into buffer 0 (each thread moves 2 x 16B of K and of V)
    *(v8h*)&ksm[0][tid * 8] = *(const v8h*)(kbh + tid * 8);
    *(v8h*)&ksm[0][s1i * 8] = *(const v8h*)(kbh + s1i * 8);
    *(v8h*)&vsm[0][(tid >> 3) * 64 + (tid & 7) * 8] = *(const v8h*)(vbhT + (size_t)(tid >> 3) * LL + (tid & 7) * 8);
    *(v8h*)&vsm[0][(s1i >> 3) * 64 + (s1i & 7) * 8] = *(const v8h*)(vbhT + (size_t)(s1i >> 3) * LL + (s1i & 7) * 8);
    __syncthreads();

    for (int c = 0; c < 16; ++c) {       // 16 chunks of 64 keys
        int cur = c & 1;
        bool more = (c + 1) < 16;
        v8h kr0, kr1, vr0, vr1;
        if (more) {                      // issue next chunk's global loads before compute
            const _Float16* kg = kbh + (size_t)(c + 1) * 64 * DD;
            kr0 = *(const v8h*)(kg + tid * 8);
            kr1 = *(const v8h*)(kg + s1i * 8);
            vr0 = *(const v8h*)(vbhT + (size_t)(tid >> 3) * LL + (c + 1) * 64 + (tid & 7) * 8);
            vr1 = *(const v8h*)(vbhT + (size_t)(s1i >> 3) * LL + (c + 1) * 64 + (s1i & 7) * 8);
        }
        if (c + 2 < 16)
            __builtin_prefetch(kbh + (size_t)(c + 2) * 64 * DD + tid * 32, 0, 2);

        const _Float16* ks = ksm[cur];
        const _Float16* vs = vsm[cur];
#pragma unroll
        for (int sub = 0; sub < 64; sub += 32) {
            int kabs = c * 64 + sub;
            // K A-fragments from LDS, shared by both q-tiles
            v16h ka00 = load_a_f16(ks + (sub) * DD,            DD, lane);
            v16h ka01 = load_a_f16(ks + (sub) * DD + 32,       DD, lane);
            v16h ka10 = load_a_f16(ks + (sub + 16) * DD,       DD, lane);
            v16h ka11 = load_a_f16(ks + (sub + 16) * DD + 32,  DD, lane);
            const float* kp = kbias + b * LL + kabs + 8 * hl;
            v8f kb0 = *(const v8f*)(kp);
            v8f kb1 = *(const v8f*)(kp + 16);

            v16h pa[2];
#pragma unroll
            for (int qi = 0; qi < 2; ++qi) {
                v8f s0 = {}, s1 = {};
                s0 = wmma_f16(ka00, bq[qi][0], s0);
                s0 = wmma_f16(ka01, bq[qi][1], s0);
                s1 = wmma_f16(ka10, bq[qi][0], s1);
                s1 = wmma_f16(ka11, bq[qi][1], s1);

                float st0[8], st1[8];
                float bmax = -3.0e38f;
#pragma unroll
                for (int r = 0; r < 8; ++r) {
                    st0[r] = s0[r] * scale + kb0[r];
                    st1[r] = s1[r] * scale + kb1[r];
                    bmax = fmaxf(bmax, fmaxf(st0[r], st1[r]));
                }
                bmax = fmaxf(bmax, __shfl_xor(bmax, 16, 32));
                float m_new = fmaxf(m_run[qi], bmax);
                float alpha = __expf(m_run[qi] - m_new);
                float psum = 0.0f;
#pragma unroll
                for (int j = 0; j < 4; ++j) {
                    float p0 = __expf(st0[2 * j]     - m_new);
                    float p1 = __expf(st0[2 * j + 1] - m_new);
                    float p2 = __expf(st1[2 * j]     - m_new);
                    float p3 = __expf(st1[2 * j + 1] - m_new);
                    psum += p0 + p1 + p2 + p3;
                    pa[qi][2 * j]           = (_Float16)p0;
                    pa[qi][2 * j + 1]       = (_Float16)p1;
                    pa[qi][2 * (j + 4)]     = (_Float16)p2;
                    pa[qi][2 * (j + 4) + 1] = (_Float16)p3;
                }
                psum += __shfl_xor(psum, 16, 32);
                l_run[qi] = l_run[qi] * alpha + psum;
                m_run[qi] = m_new;

                float af[8];
#pragma unroll
                for (int r = 0; r < 8; ++r) af[r] = __shfl(alpha, r + 8 * hl, 32);
#pragma unroll
                for (int dt = 0; dt < 4; ++dt)
#pragma unroll
                    for (int r = 0; r < 8; ++r) o[qi][dt][r] *= af[r];
            }
            // V^T B-fragments from LDS, each used by both q-tiles
#pragma unroll
            for (int dt = 0; dt < 4; ++dt) {
                v16h vb = load_b_colT(vs + (dt * 16) * 64 + sub, 64, lane);
                o[0][dt] = wmma_f16(pa[0], vb, o[0][dt]);
                o[1][dt] = wmma_f16(pa[1], vb, o[1][dt]);
            }
        }
        if (more) {
            int nb = 1 - cur;
            *(v8h*)&ksm[nb][tid * 8] = kr0;
            *(v8h*)&ksm[nb][s1i * 8] = kr1;
            *(v8h*)&vsm[nb][(tid >> 3) * 64 + (tid & 7) * 8] = vr0;
            *(v8h*)&vsm[nb][(s1i >> 3) * 64 + (s1i & 7) * 8] = vr1;
        }
        __syncthreads();
    }

    // finalize both q-tiles: divide by row sum, zero masked queries; store [B,L,H,D] = [B*L,E]
#pragma unroll
    for (int qi = 0; qi < 2; ++qi) {
        float qm  = (QMask[b * LL + q0[qi] + n] != 0) ? 1.0f : 0.0f;
        float fac = qm / l_run[qi];
        float fr[8];
#pragma unroll
        for (int r = 0; r < 8; ++r) fr[r] = __shfl(fac, r + 8 * hl, 32);
#pragma unroll
        for (int dt = 0; dt < 4; ++dt)
#pragma unroll
            for (int r = 0; r < 8; ++r) {
                int q = r + 8 * hl;
                o16[((size_t)(b * LL + q0[qi] + q) * HH + h) * DD + dt * 16 + n] =
                    (_Float16)(o[qi][dt][r] * fr[r]);
            }
    }
}

// ---- kernel 3: output projection, 64x64 tile per wave (16 WMMAs per 32-wide k-step)
__global__ __launch_bounds__(256) void mha_outproj_kernel(
    const _Float16* __restrict__ o16, const _Float16* __restrict__ woT, float* __restrict__ out) {
    int lane = threadIdx.x & 31;
    int wid  = blockIdx.x * 8 + (threadIdx.x >> 5);
    int m0 = (wid >> 4) * 64, n0 = (wid & 15) * 64;
    v8f c[4][4] = {};
    for (int kk = 0; kk < EE; kk += 32) {
        v16h a[4];
#pragma unroll
        for (int mi = 0; mi < 4; ++mi)
            a[mi] = load_a_f16(o16 + (size_t)(m0 + mi * 16) * EE + kk, EE, lane);
#pragma unroll
        for (int ni = 0; ni < 4; ++ni) {
            v16h bb = load_b_colT(woT + (size_t)(n0 + ni * 16) * EE + kk, EE, lane);
#pragma unroll
            for (int mi = 0; mi < 4; ++mi)
                c[mi][ni] = wmma_f16(a[mi], bb, c[mi][ni]);
        }
    }
    int n = lane & 15, hl = lane >> 4;
#pragma unroll
    for (int mi = 0; mi < 4; ++mi)
#pragma unroll
        for (int ni = 0; ni < 4; ++ni)
#pragma unroll
            for (int r = 0; r < 8; ++r)
                out[(size_t)(m0 + mi * 16 + r + 8 * hl) * EE + n0 + ni * 16 + n] = c[mi][ni][r];
}

extern "C" void kernel_launch(void* const* d_in, const int* in_sizes, int n_in,
                              void* d_out, int out_size, void* d_ws, size_t ws_size,
                              hipStream_t stream) {
    const float* Q  = (const float*)d_in[0];
    const float* K  = (const float*)d_in[1];
    const float* V  = (const float*)d_in[2];
    const float* HL = (const float*)d_in[3];
    const float* WO = (const float*)d_in[4];
    const int* QMask = (const int*)d_in[5];
    const int* KMask = (const int*)d_in[6];
    float* out = (float*)d_out;

    char* ws = (char*)d_ws;
    const size_t MB = 1u << 20;
    _Float16* qh    = (_Float16*)(ws + 0 * MB);   // 16 MB  [B,H,L,D]
    _Float16* kh    = (_Float16*)(ws + 16 * MB);  // 16 MB  [B,H,L,D]
    _Float16* vhT   = (_Float16*)(ws + 32 * MB);  // 16 MB  [B,H,D,L]
    _Float16* o16   = (_Float16*)(ws + 48 * MB);  // 16 MB  [B,L,H,D]
    _Float16* whT   = (_Float16*)(ws + 64 * MB);  // 128 KB
    _Float16* woT   = (_Float16*)(ws + 65 * MB);  // 2 MB
    float*    kbias = (float*)   (ws + 67 * MB);  // 32 KB

    mha_prep_kernel<<<(EE * EE + 255) / 256, 256, 0, stream>>>(HL, WO, KMask, whT, woT, kbias);
    mha_proj_kernel<<<3072, 256, 0, stream>>>(Q, K, V, whT, qh, kh, vhT);      // 24576 waves
    mha_attn_kernel<<<512, 256, 0, stream>>>(qh, kh, vhT, kbias, QMask, o16);  // 8 waves/block, LDS staged
    mha_outproj_kernel<<<256, 256, 0, stream>>>(o16, woT, out);                // 2048 waves, 64x64 tiles
}
